// Updating_memory_5282809774416
// MI455X (gfx1250) — matrix operand order repaired and checked by reading, not compiled, not run on hardware
//
#include <hip/hip_runtime.h>
#include <math.h>

typedef __attribute__((ext_vector_type(8)))  __bf16 v8bf;
typedef __attribute__((ext_vector_type(16))) __bf16 v16bf;
typedef __attribute__((ext_vector_type(8)))  float  v8f;

#define L 5
#define H 100
#define NB 16                // batches per workgroup
#define ROWS (NB * L)        // 80 rows = 5 exact M-tiles
#define MT 5
#define NT 7                 // 7*16 = 112 >= 100
#define NTILES (MT * NT)     // 35 C-tiles per GEMM
#define KP 128               // K padded 100 -> 128 (4 steps of 32)
#define WN 112               // padded N for the weight panel
#define NWAVES 8
#define THREADS 256
#define TPW 5                // max tiles per wave: ceil(35/8)

#define LOG2E 1.4426950408889634f

// branch-free fast transcendentals: single v_exp_f32 / v_rcp_f32 TRANS ops
__device__ __forceinline__ float fast_exp(float x) {
    return __builtin_amdgcn_exp2f(x * LOG2E);
}
__device__ __forceinline__ float fast_sigmoid(float x) {
    return __builtin_amdgcn_rcpf(1.f + __builtin_amdgcn_exp2f(-x * LOG2E));
}
__device__ __forceinline__ float fast_tanh(float x) {
    // tanh(x) = 1 - 2/(exp(2x)+1)
    return 1.f - 2.f * __builtin_amdgcn_rcpf(__builtin_amdgcn_exp2f(x * (2.f * LOG2E)) + 1.f);
}

// A-fragment (16x32 bf16): lane holds K = 16*hi..16*hi+15 of its row
__device__ __forceinline__ v16bf load_afrag(const __bf16* rowp, int hi) {
    union { v16bf v; v8bf h[2]; } u;
    u.h[0] = *(const v8bf*)(rowp + 16 * hi);
    u.h[1] = *(const v8bf*)(rowp + 16 * hi + 8);
    return u.v;
}

__global__ __launch_bounds__(THREADS, 1)
void fused_update_memory(
    const float* __restrict__ his_emb, const float* __restrict__ sub_emb,
    const float* __restrict__ memory,
    const int* __restrict__ sub_len, const int* __restrict__ his_len,
    const float* __restrict__ W_his, const float* __restrict__ b_his,
    const float* __restrict__ W_sub, const float* __restrict__ b_sub,
    const float* __restrict__ W_mem, const float* __restrict__ b_mem,
    const float* __restrict__ W_gate, const float* __restrict__ b_gate,
    float* __restrict__ out, int B)
{
    // bf16 operand staging (fp32 accumulate in WMMA)
    __shared__ __align__(32) __bf16 sEmbH[3][ROWS * KP];  // his/sub/mem, K-padded w/ zeros (61440 B)
    __shared__ __align__(32) __bf16 sTanH[3][ROWS * KP];  // activations; [0]->mem_att, [1]->mem_att*mem (61440 B)
    __shared__ __align__(32) __bf16 sWH[WN * KP];         // weight panel, TRANSPOSED [N][K]   (28672 B)
    __shared__ float sS[NB][2][L][L];                     // attention scores/weights          (3200 B)
    __shared__ float sBias[4][H];                         // b_his/b_sub/b_mem/b_gate          (1600 B)

    const int tid  = threadIdx.x;
    const int wave = tid >> 5;
    const int lane = tid & 31;
    const int lo   = lane & 15;
    const int hi   = lane >> 4;
    const int wg   = blockIdx.x;
    const long long base = (long long)wg * (NB * L * H);

    // Warm L2 with the gate weights (used last): global_prefetch_b8 path
    for (int off = tid * 256; off < 3 * H * H * 4; off += THREADS * 256)
        __builtin_prefetch((const char*)W_gate + off, 0, 0);

    // ------------------ stage biases in LDS ------------------
    {
        const float* bsrc[4] = { b_his, b_sub, b_mem, b_gate };
        for (int idx = tid; idx < 4 * H; idx += THREADS)
            sBias[idx / H][idx % H] = bsrc[idx / H][idx % H];
    }

    // ------------------ load embeddings -> bf16 LDS (zero K-pad) ------------------
    {
        const float* srcs[3] = { his_emb, sub_emb, memory };
        #pragma unroll
        for (int m = 0; m < 3; ++m) {
            for (int idx = tid; idx < ROWS * KP; idx += THREADS) {
                int r = idx / KP, d = idx % KP;
                float v = 0.f;
                if (d < H && (wg * NB + r / L) < B) v = srcs[m][base + r * H + d];
                sEmbH[m][idx] = (__bf16)v;
            }
        }
    }

    // ------------------ phase 1: three tanh GEMMs (bf16 WMMA, fp32 acc) ------------------
    {
        const float* Ws[3] = { W_his, W_sub, W_mem };
        #pragma unroll 1
        for (int mat = 0; mat < 3; ++mat) {
            __syncthreads();   // previous sW consumers done (covers emb/bias load on mat==0)
            // coalesced global read, transposed bf16 store: sWH[n][k] = W[k][n]
            for (int idx = tid; idx < H * WN; idx += THREADS) {
                int k = idx / WN, n = idx % WN;
                float v = (n < H) ? Ws[mat][k * H + n] : 0.f;
                sWH[n * KP + k] = (__bf16)v;
            }
            for (int idx = tid; idx < WN * (KP - H); idx += THREADS) {   // zero K-pad rows
                int n = idx / (KP - H), k = H + idx % (KP - H);
                sWH[n * KP + k] = (__bf16)0.f;
            }
            __syncthreads();

            for (int t = wave; t < NTILES; t += NWAVES) {   // wave-uniform
                const int mrow0 = (t / NT) * 16;
                const int n0    = (t % NT) * 16;
                v8f c = {};
                #pragma unroll
                for (int kk = 0; kk < KP; kk += 32) {
                    v16bf af = load_afrag(&sEmbH[mat][(mrow0 + lo) * KP + kk], hi);
                    v16bf bf = *(const v16bf*)(&sWH[(n0 + lo) * KP + kk + 16 * hi]);
                    c = __builtin_amdgcn_wmma_f32_16x16x32_bf16(
                            false, af, false, bf, (short)0, c, false, false);
                }
                const int n = n0 + lo;
                if (n < H) {
                    float bb = sBias[mat][n];
                    #pragma unroll
                    for (int r = 0; r < 8; ++r) {
                        int row = mrow0 + r + 8 * hi;
                        sTanH[mat][row * KP + n] = (__bf16)fast_tanh(c[r] + bb);
                    }
                }
            }
        }
    }
    __syncthreads();

    // ------------------ phase 2a: 5x5 attention scores ------------------
    for (int p = tid; p < NB * 2 * L * L; p += THREADS) {
        int b = p / 50, q = p % 50;
        int which = q / 25, idx = q % 25;
        int m = idx / L, j = idx % L;
        const __bf16* ra = &sTanH[2][(b * L + m) * KP];       // mem_4att row
        const __bf16* rb = &sTanH[which][(b * L + j) * KP];   // his/sub _4att row
        float acc = 0.f;
        for (int d = 0; d < H; ++d) acc += (float)ra[d] * (float)rb[d];
        sS[b][which][m][j] = acc;
    }
    __syncthreads();

    // ------------------ phase 2b: softmax over j; zero rows with m >= len ------------------
    if (tid < NB * 2 * L) {
        int b = tid / (2 * L), q = tid % (2 * L);
        int which = q / L, m = q % L;
        int gb = wg * NB + b;
        int len = 0;
        if (gb < B) len = which ? sub_len[gb] : his_len[gb];
        float* row = sS[b][which][m];
        float mx = row[0];
        #pragma unroll
        for (int j = 1; j < L; ++j) mx = fmaxf(mx, row[j]);
        float e[L], sum = 0.f;
        #pragma unroll
        for (int j = 0; j < L; ++j) { e[j] = fast_exp(row[j] - mx); sum += e[j]; }
        float scale = (m < len) ? __builtin_amdgcn_rcpf(sum) : 0.f;
        #pragma unroll
        for (int j = 0; j < L; ++j) row[j] = e[j] * scale;
    }
    __syncthreads();

    // ------- phase 2c: weighted sums -> mem_att (sTanH[0]) and mem_att*memory (sTanH[1]) -------
    __bf16* sAtt  = sTanH[0];
    __bf16* sProd = sTanH[1];
    for (int idx = tid; idx < ROWS * KP; idx += THREADS) {
        int r = idx / KP, d = idx % KP;
        int b = r / L, m = r % L;
        float v = 0.f;
        if (d < H) {
            #pragma unroll
            for (int j = 0; j < L; ++j) {
                v += sS[b][0][m][j] * (float)sEmbH[0][(b * L + j) * KP + d];
                v += sS[b][1][m][j] * (float)sEmbH[1][(b * L + j) * KP + d];
            }
        }
        float mem = (float)sEmbH[2][idx];
        sAtt[idx]  = (__bf16)v;
        sProd[idx] = (__bf16)(v * mem);
    }

    // ------------------ phase 3: gate GEMM, K=300 in 3 chunks of 100 ------------------
    v8f accs[TPW] = {};
    #pragma unroll 1
    for (int chunk = 0; chunk < 3; ++chunk) {
        __syncthreads();   // sW consumers done; fences phase-2c writes on chunk 0
        for (int idx = tid; idx < H * WN; idx += THREADS) {
            int k = idx / WN, n = idx % WN;
            float v = (n < H) ? W_gate[(chunk * H + k) * H + n] : 0.f;
            sWH[n * KP + k] = (__bf16)v;
        }
        for (int idx = tid; idx < WN * (KP - H); idx += THREADS) {
            int n = idx / (KP - H), k = H + idx % (KP - H);
            sWH[n * KP + k] = (__bf16)0.f;
        }
        __syncthreads();

        const __bf16* Asrc = (chunk == 0) ? sAtt : (chunk == 1) ? sEmbH[2] : sProd;
        #pragma unroll
        for (int i = 0; i < TPW; ++i) {
            const int t = wave + i * NWAVES;
            if (t < NTILES) {   // wave-uniform branch: EXEC stays all-ones
                const int mrow0 = (t / NT) * 16;
                const int n0    = (t % NT) * 16;
                v8f c = accs[i];
                #pragma unroll
                for (int kk = 0; kk < KP; kk += 32) {
                    v16bf af = load_afrag(&Asrc[(mrow0 + lo) * KP + kk], hi);
                    v16bf bf = *(const v16bf*)(&sWH[(n0 + lo) * KP + kk + 16 * hi]);
                    c = __builtin_amdgcn_wmma_f32_16x16x32_bf16(
                            false, af, false, bf, (short)0, c, false, false);
                }
                accs[i] = c;
            }
        }
    }

    // ------------------ epilogue: sigmoid gate + blend, store fp32 ------------------
    #pragma unroll
    for (int i = 0; i < TPW; ++i) {
        const int t = wave + i * NWAVES;
        if (t < NTILES) {
            const int mrow0 = (t / NT) * 16;
            const int n0    = (t % NT) * 16;
            const int n     = n0 + lo;
            if (n < H) {
                float bg = sBias[3][n];
                v8f c = accs[i];
                #pragma unroll
                for (int r = 0; r < 8; ++r) {
                    int row = mrow0 + r + 8 * hi;
                    if ((wg * NB + row / L) < B) {
                        float mem  = (float)sEmbH[2][row * KP + n];
                        float matt = (float)sAtt[row * KP + n];
                        float g = fast_sigmoid(c[r] + bg);
                        out[base + row * H + n] = (1.f - g) * mem + g * matt;
                    }
                }
            }
        }
    }
}

extern "C" void kernel_launch(void* const* d_in, const int* in_sizes, int n_in,
                              void* d_out, int out_size, void* d_ws, size_t ws_size,
                              hipStream_t stream) {
    const float* his_emb = (const float*)d_in[0];
    const float* sub_emb = (const float*)d_in[1];
    const float* memory  = (const float*)d_in[2];
    const int*   sub_len = (const int*)d_in[3];
    const int*   his_len = (const int*)d_in[4];
    const float* W_his   = (const float*)d_in[5];
    const float* b_his   = (const float*)d_in[6];
    const float* W_sub   = (const float*)d_in[7];
    const float* b_sub   = (const float*)d_in[8];
    const float* W_mem   = (const float*)d_in[9];
    const float* b_mem   = (const float*)d_in[10];
    const float* W_gate  = (const float*)d_in[11];
    const float* b_gate  = (const float*)d_in[12];
    float* out = (float*)d_out;

    const int B = in_sizes[0] / (L * H);
    const int grid = (B + NB - 1) / NB;
    fused_update_memory<<<grid, THREADS, 0, stream>>>(
        his_emb, sub_emb, memory, sub_len, his_len,
        W_his, b_his, W_sub, b_sub, W_mem, b_mem, W_gate, b_gate,
        out, B);
}
